// MyTRN_22119081574971
// MI455X (gfx1250) — compile-verified, matrix operand order checked by reading, not run on hardware
//
#include <hip/hip_runtime.h>

typedef __attribute__((ext_vector_type(16))) _Float16 v16h;
typedef __attribute__((ext_vector_type(8)))  _Float16 v8h;
typedef __attribute__((ext_vector_type(8)))  float    v8f;

#define BM 128
#define BN 128
#define BK 32
#define LDSS 40   // halves per LDS row: 32 + 8 pad (80B, keeps ds_load_b128 16B-aligned)

// matches the builtin's parameter type exactly (per hipcc diagnostic):
// '__attribute__((__vector_size__(4 * sizeof(int)))) int __device__ *'
typedef int v4i __attribute__((vector_size(4 * sizeof(int))));
typedef __attribute__((address_space(1))) v4i* g4p_t;   // global (__device__) int4*
typedef __attribute__((address_space(3))) v4i* l4p_t;   // LDS (__shared__) int4*

// 16B global -> LDS async copy (ASYNCcnt-tracked)
__device__ __forceinline__ void async_copy16(const void* g, void* l) {
#if __has_builtin(__builtin_amdgcn_global_load_async_to_lds_b128)
    __builtin_amdgcn_global_load_async_to_lds_b128(
        (g4p_t)(uintptr_t)g,                            // strip const, AS1
        (l4p_t)(uintptr_t)(uint32_t)(uintptr_t)l,       // low 32 bits = LDS offset
        0, 0);
#else
    asm volatile("global_load_async_to_lds_b128 %0, %1, off"
                 :: "v"((uint32_t)(uintptr_t)l), "v"((uint64_t)(uintptr_t)g)
                 : "memory");
#endif
}

template<int N>
__device__ __forceinline__ void wait_async() {
#if __has_builtin(__builtin_amdgcn_s_wait_asynccnt)
    __builtin_amdgcn_s_wait_asynccnt(N);
#else
    asm volatile("s_wait_asynccnt %0" :: "i"(N) : "memory");
#endif
}

// C[N, M] = X1[N,K1] @ W1[M,K1]^T  (+ X2[N,K2] @ W2[M,K2]^T if K2>0)
// f32 accumulate via WMMA f16, register-fused dual-GEMM, async double-buffered LDS.
//
// Epilogue modes:
//  - GATES: weights are gate-packed (row' = 4h+gate, order i,f,g,o). Each
//    lane's 8-value strip = full gate sets for h0, h0+1 -> LSTM cell update
//    done in-register: c_out = sig(f)*c_in + sig(i)*tanh(g),
//    h_out = sig(o)*tanh(c_out). bias0 = packed (bih+bhh). No G buffer.
//  - else: optional bias0+bias1 (per-M), optional ReLU; vectorized f32
//    (row stride outF_rs) and/or f16 (row stride M) stores.
template<bool B_IS_F32, bool RELU, bool GATES>
__global__ void __launch_bounds__(256)
gemm2_wmma_kernel(const _Float16* __restrict__ W1, const void* __restrict__ X1, int K1,
                  const _Float16* __restrict__ W2, const _Float16* __restrict__ X2, int K2,
                  const float* __restrict__ bias0, const float* __restrict__ bias1,
                  float* __restrict__ outF, long outF_rs,
                  _Float16* __restrict__ outH,
                  const float* __restrict__ c_in, float* __restrict__ c_out,
                  _Float16* __restrict__ h_out,
                  int M, int N)
{
    __shared__ _Float16 lA[2][BM * LDSS];   // weights tile, double buffered
    __shared__ _Float16 lB[2][BN * LDSS];   // activations tile, double buffered

    const int t    = threadIdx.x;
    const int wid  = t >> 5;
    const int lane = t & 31;

    const int lrow = t >> 1;            // staging row 0..127
    const int lcol = (t & 1) << 4;      // 0 or 16 halves (32B per thread per matrix)

    const int wmBase = (wid & 3) * 32;  // wave M offset in tile
    const int wnBase = (wid >> 2) * 64; // wave N offset in tile
    const int lrow16 = lane & 15;
    const int lhalf  = ((lane >> 4) & 1) * 8;

    const int m0 = blockIdx.y * BM;
    const int n0 = blockIdx.x * BN;

    const int t1    = K1 >> 5;           // tiles in segment 1
    const int tiles = (K1 + K2) >> 5;    // total K tiles over both segments

    auto stage = [&](int ti, int buf) {
        _Float16* la = &lA[buf][lrow * LDSS + lcol];
        _Float16* lb = &lB[buf][lrow * LDSS + lcol];
        if (ti < t1) {
            const int kt = ti << 5;
            const _Float16* ga = W1 + (size_t)(m0 + lrow) * K1 + kt + lcol;
            async_copy16(ga,     la);
            async_copy16(ga + 8, la + 8);
            if (B_IS_F32) {
                // f32 source needs conversion: synchronous load + ds_store
                const float* gx = (const float*)X1 + (size_t)(n0 + lrow) * K1 + kt + lcol;
#pragma unroll
                for (int i = 0; i < 4; ++i) {
                    float4 v = ((const float4*)gx)[i];
                    lb[4*i+0] = (_Float16)v.x; lb[4*i+1] = (_Float16)v.y;
                    lb[4*i+2] = (_Float16)v.z; lb[4*i+3] = (_Float16)v.w;
                }
            } else {
                const _Float16* gb = (const _Float16*)X1 + (size_t)(n0 + lrow) * K1 + kt + lcol;
                async_copy16(gb,     lb);
                async_copy16(gb + 8, lb + 8);
            }
        } else {
            const int kt = (ti - t1) << 5;
            const _Float16* ga = W2 + (size_t)(m0 + lrow) * K2 + kt + lcol;
            const _Float16* gb = X2 + (size_t)(n0 + lrow) * K2 + kt + lcol;
            async_copy16(ga,     la);
            async_copy16(ga + 8, la + 8);
            async_copy16(gb,     lb);
            async_copy16(gb + 8, lb + 8);
        }
    };

    v8f acc[2][4];
    const v8f vzero = {0.f, 0.f, 0.f, 0.f, 0.f, 0.f, 0.f, 0.f};
#pragma unroll
    for (int mt = 0; mt < 2; ++mt)
#pragma unroll
        for (int nt = 0; nt < 4; ++nt)
            acc[mt][nt] = vzero;

    // each wave issues 4 async b128 per tile (2 if B side is synchronous f32)
    constexpr int THRESH = B_IS_F32 ? 2 : 4;

    stage(0, 0);
    for (int ti = 0; ti < tiles; ++ti) {
        const int buf = ti & 1;
        if (ti + 1 < tiles) {
            stage(ti + 1, buf ^ 1);      // prefetch next tile into other buffer
            wait_async<THRESH>();        // <=THRESH outstanding => tile ti landed
        } else {
            wait_async<0>();
        }
        __syncthreads();

        v16h a[2], b[4];
#pragma unroll
        for (int mt = 0; mt < 2; ++mt) {
            const _Float16* p = &lA[buf][(wmBase + mt*16 + lrow16) * LDSS + lhalf];
            v8h lo = *(const v8h*)p;
            v8h hi = *(const v8h*)(p + 16);
            a[mt] = __builtin_shufflevector(lo, hi, 0,1,2,3,4,5,6,7,8,9,10,11,12,13,14,15);
        }
#pragma unroll
        for (int nt = 0; nt < 4; ++nt) {
            const _Float16* p = &lB[buf][(wnBase + nt*16 + lrow16) * LDSS + lhalf];
            v8h lo = *(const v8h*)p;
            v8h hi = *(const v8h*)(p + 16);
            b[nt] = __builtin_shufflevector(lo, hi, 0,1,2,3,4,5,6,7,8,9,10,11,12,13,14,15);
        }
#pragma unroll
        for (int mt = 0; mt < 2; ++mt)
#pragma unroll
            for (int nt = 0; nt < 4; ++nt)
                acc[mt][nt] = __builtin_amdgcn_wmma_f32_16x16x32_f16(
                    false, a[mt], false, b[nt], (short)0, acc[mt][nt], false, false);

        __syncthreads();   // protect buffer reuse by next stage()
    }

    // ---- epilogue ----
    const int gm = m0 + wmBase;
    const int gn = n0 + wnBase;
    if (GATES) {
        const int Hh = M >> 2;      // hidden size (M = 4H gate-packed rows)
#pragma unroll
        for (int mt = 0; mt < 2; ++mt) {
#pragma unroll
            for (int nt = 0; nt < 4; ++nt) {
                const int n  = gn + nt*16 + lrow16;
                const int mb = gm + mt*16 + lhalf;   // multiple of 8 => gate-aligned
#pragma unroll
                for (int j = 0; j < 2; ++j) {
                    float gi = acc[mt][nt][4*j+0] + bias0[mb + 4*j + 0];
                    float gf = acc[mt][nt][4*j+1] + bias0[mb + 4*j + 1];
                    float gg = acc[mt][nt][4*j+2] + bias0[mb + 4*j + 2];
                    float go = acc[mt][nt][4*j+3] + bias0[mb + 4*j + 3];
                    const int h = (mb >> 2) + j;
                    const size_t hi = (size_t)n * Hh + h;
                    float cp = c_in ? c_in[hi] : 0.f;
                    float si = 1.f / (1.f + __expf(-gi));
                    float sf = 1.f / (1.f + __expf(-gf));
                    float so = 1.f / (1.f + __expf(-go));
                    float cn = sf * cp + si * tanhf(gg);
                    c_out[hi] = cn;
                    h_out[hi] = (_Float16)(so * tanhf(cn));
                }
            }
        }
    } else {
#pragma unroll
        for (int mt = 0; mt < 2; ++mt) {
#pragma unroll
            for (int nt = 0; nt < 4; ++nt) {
                const int n  = gn + nt*16 + lrow16;
                const int mb = gm + mt*16 + lhalf;   // multiple of 8 => 16B aligned
                float vals[8];
#pragma unroll
                for (int v = 0; v < 8; ++v) {
                    float val = acc[mt][nt][v];
                    if (bias0) val += bias0[mb + v];
                    if (bias1) val += bias1[mb + v];
                    if (RELU)  val = fmaxf(val, 0.f);
                    vals[v] = val;
                }
                if (outF) {
                    float4* p = (float4*)(outF + (size_t)n * outF_rs + mb);
                    p[0] = float4{vals[0], vals[1], vals[2], vals[3]};
                    p[1] = float4{vals[4], vals[5], vals[6], vals[7]};
                }
                if (outH) {
                    v8h hv;
#pragma unroll
                    for (int v = 0; v < 8; ++v) hv[v] = (_Float16)vals[v];
                    *(v8h*)(outH + (size_t)n * M + mb) = hv;
                }
            }
        }
    }
}

__global__ void cvt_f32_f16_kernel(const float* __restrict__ in,
                                   _Float16* __restrict__ out, long n)
{
    long i = (long)blockIdx.x * blockDim.x + threadIdx.x;
    if (i < n) out[i] = (_Float16)in[i];
}

// repack LSTM weight rows: src row = gate*H + h  ->  dst row = 4*h + gate (f16)
__global__ void pack_lstm_w_kernel(const float* __restrict__ src,
                                   _Float16* __restrict__ dst, int H, int K)
{
    long idx = (long)blockIdx.x * blockDim.x + threadIdx.x;
    if (idx >= (long)4 * H * K) return;
    int k    = (int)(idx % K);
    int row  = (int)(idx / K);
    int gate = row / H, h = row % H;
    dst[((size_t)4 * h + gate) * K + k] = (_Float16)src[idx];
}

// packed bias sum: dst[4h+gate] = b0[gate*H+h] + b1[gate*H+h]
__global__ void pack_bias_kernel(const float* __restrict__ b0,
                                 const float* __restrict__ b1,
                                 float* __restrict__ dst, int H)
{
    int idx = blockIdx.x * blockDim.x + threadIdx.x;
    if (idx >= 4 * H) return;
    int gate = idx / H, h = idx % H;
    dst[4 * h + gate] = b0[idx] + b1[idx];
}

__global__ void scale_cvt_kernel(const float* __restrict__ in, long rstride,
                                 float scale, _Float16* __restrict__ out,
                                 int N, int F)
{
    long idx = (long)blockIdx.x * blockDim.x + threadIdx.x;
    if (idx >= (long)N * F) return;
    int n = (int)(idx / F), f = (int)(idx % F);
    out[idx] = (_Float16)(in[(size_t)n * rstride + f] * scale);
}

// enc_scores[n, c] = h[n,:] . Wec[c,:] + bec[c]   (tiny: C=22)
__global__ void proj_kernel(const _Float16* __restrict__ h,
                            const float* __restrict__ Wc,
                            const float* __restrict__ bc,
                            float* __restrict__ out, int N, int C, int F)
{
    int idx = blockIdx.x * blockDim.x + threadIdx.x;
    if (idx >= N * C) return;
    int n = idx / C, c = idx % C;
    const _Float16* hr = h + (size_t)n * F;
    const float*    wr = Wc + (size_t)c * F;
    float s = bc[c];
    for (int k = 0; k < F; ++k) s += (float)hr[k] * wr[k];
    out[idx] = s;
}

extern "C" void kernel_launch(void* const* d_in, const int* in_sizes, int n_in,
                              void* d_out, int out_size, void* d_ws, size_t ws_size,
                              hipStream_t stream)
{
    (void)in_sizes; (void)n_in; (void)out_size; (void)ws_size;

    const float* x    = (const float*)d_in[0];
    const float* Wf   = (const float*)d_in[1];
    const float* bf   = (const float*)d_in[2];
    const float* dWih = (const float*)d_in[3];
    const float* dWhh = (const float*)d_in[4];
    const float* dbih = (const float*)d_in[5];
    const float* dbhh = (const float*)d_in[6];
    const float* Wdc  = (const float*)d_in[7];
    const float* bdc  = (const float*)d_in[8];
    const float* eWih = (const float*)d_in[9];
    const float* eWhh = (const float*)d_in[10];
    const float* ebih = (const float*)d_in[11];
    const float* ebhh = (const float*)d_in[12];
    const float* Wec  = (const float*)d_in[13];
    const float* bec  = (const float*)d_in[14];
    float* out = (float*)d_out;

    constexpr int Bb = 64, Tt = 64, Dd = 8, F = 1024, H = 1024, C = 22, IN = 4096;
    constexpr int N = Bb * Tt;  // 4096

    char* ws = (char*)d_ws;
    size_t off = 0;
    auto take = [&](size_t bytes) {
        char* p = ws + off;
        off += (bytes + 255) & ~(size_t)255;
        return p;
    };
    _Float16* wf_h   = (_Float16*)take((size_t)F * IN * 2);
    _Float16* wih_p  = (_Float16*)take((size_t)4 * H * F * 2);   // gate-packed
    _Float16* whh_p  = (_Float16*)take((size_t)4 * H * H * 2);   // gate-packed
    _Float16* wdc_h  = (_Float16*)take((size_t)F * H * 2);
    _Float16* ewih_p = (_Float16*)take((size_t)4 * F * F * 2);   // gate-packed
    _Float16* ewhh_p = (_Float16*)take((size_t)4 * F * F * 2);   // gate-packed
    float*    bdec_p = (float*)   take((size_t)4 * H * 4);       // packed bih+bhh
    float*    benc_p = (float*)   take((size_t)4 * F * 4);
    _Float16* feats_h= (_Float16*)take((size_t)N * F * 2);
    _Float16* h_h    = (_Float16*)take((size_t)N * H * 2);
    float*    cbuf   = (float*)   take((size_t)N * H * 4);
    _Float16* out_h  = (_Float16*)take((size_t)N * F * 2);
    _Float16* h0_h   = (_Float16*)take((size_t)N * F * 2);
    _Float16* ench_h = (_Float16*)take((size_t)N * F * 2);

    // weight conversion / gate-packing
    cvt_f32_f16_kernel<<<dim3((unsigned)(((long)F*IN + 255)/256)), dim3(256), 0, stream>>>(
        Wf, wf_h, (long)F*IN);
    cvt_f32_f16_kernel<<<dim3((unsigned)(((long)F*H + 255)/256)), dim3(256), 0, stream>>>(
        Wdc, wdc_h, (long)F*H);
    pack_lstm_w_kernel<<<dim3((unsigned)(((long)4*H*F + 255)/256)), dim3(256), 0, stream>>>(
        dWih, wih_p, H, F);
    pack_lstm_w_kernel<<<dim3((unsigned)(((long)4*H*H + 255)/256)), dim3(256), 0, stream>>>(
        dWhh, whh_p, H, H);
    pack_lstm_w_kernel<<<dim3((unsigned)(((long)4*F*F + 255)/256)), dim3(256), 0, stream>>>(
        eWih, ewih_p, F, F);
    pack_lstm_w_kernel<<<dim3((unsigned)(((long)4*F*F + 255)/256)), dim3(256), 0, stream>>>(
        eWhh, ewhh_p, F, F);
    pack_bias_kernel<<<dim3((4*H + 255)/256), dim3(256), 0, stream>>>(dbih, dbhh, bdec_p, H);
    pack_bias_kernel<<<dim3((4*F + 255)/256), dim3(256), 0, stream>>>(ebih, ebhh, benc_p, F);

    // 1) feats = relu(x @ Wf^T + bf) -> f16 (f32 activation source)
    {
        dim3 g(N / BN, F / BM), blk(256);
        gemm2_wmma_kernel<true, true, false><<<g, blk, 0, stream>>>(
            wf_h, (const void*)x, IN, nullptr, nullptr, 0,
            bf, nullptr, nullptr, 0, feats_h,
            nullptr, nullptr, nullptr, F, N);
    }

    float* decO = out + (size_t)N * C;   // dec_scores base, row stride D*F

    // 2) decoder recurrence: fused (Wih+Whh) GEMM + in-register LSTM cell,
    //    then Wdc projection straight into the strided dec_scores slice.
    const _Float16* inp = feats_h;
    for (int d = 0; d < Dd; ++d) {
        dim3 gc(N / BN, (4 * H) / BM), blk(256);
        gemm2_wmma_kernel<false, false, true><<<gc, blk, 0, stream>>>(
            wih_p, (const void*)inp, F,
            whh_p, (d > 0) ? h_h : nullptr, (d > 0) ? H : 0,
            bdec_p, nullptr, nullptr, 0, nullptr,
            (d == 0) ? nullptr : cbuf, cbuf, h_h, 4 * H, N);

        dim3 go(N / BN, F / BM);
        gemm2_wmma_kernel<false, false, false><<<go, blk, 0, stream>>>(
            wdc_h, (const void*)h_h, H, nullptr, nullptr, 0,
            bdc, nullptr, decO + (size_t)d * F, (long)Dd * F, out_h,
            nullptr, nullptr, nullptr, F, N);
        inp = out_h;
    }

    // 3) encoder: h0 = last_out / D (f16), c0 = 0; fused cell GEMM
    scale_cvt_kernel<<<dim3((unsigned)(((long)N * F + 255) / 256)), dim3(256), 0, stream>>>(
        decO + (size_t)(Dd - 1) * F, (long)Dd * F, 1.0f / (float)Dd, h0_h, N, F);
    {
        dim3 gc(N / BN, (4 * F) / BM), blk(256);
        gemm2_wmma_kernel<false, false, true><<<gc, blk, 0, stream>>>(
            ewih_p, (const void*)feats_h, F, ewhh_p, h0_h, F,
            benc_p, nullptr, nullptr, 0, nullptr,
            nullptr, cbuf, ench_h, 4 * F, N);
    }

    // 4) enc_scores = enc_h @ Wec^T + bec  (tiny, scalar)
    proj_kernel<<<dim3((N * C + 255) / 256), dim3(256), 0, stream>>>(
        ench_h, Wec, bec, out, N, C, F);
}
